// DGCNNEncoder_58952721105076
// MI455X (gfx1250) — matrix operand order repaired and checked by reading, not compiled, not run on hardware
//
#include <hip/hip_runtime.h>

// ---------------------------------------------------------------------------
// DGCNN encoder for MI455X (gfx1250, wave32, WMMA).
// B=8, N=2048, K=8, IN=5, BN=16384, E=BN*K=131072.
// ---------------------------------------------------------------------------

typedef __bf16 v16bf __attribute__((ext_vector_type(16)));
typedef __bf16 v8bf  __attribute__((ext_vector_type(8)));
typedef float  v8f   __attribute__((ext_vector_type(8)));

#define BATCH 8
#define NPTS  2048
#define KNN   8
#define BNPTS (BATCH * NPTS)          // 16384
#define NEDGE (BNPTS * KNN)           // 131072
#define MTILES (NEDGE / 16)           // 8192

__device__ __forceinline__ __bf16 f2bf(float f) {
    unsigned u; __builtin_memcpy(&u, &f, 4);
    unsigned r = u + 0x7FFFu + ((u >> 16) & 1u);   // round-to-nearest-even
    unsigned short h = (unsigned short)(r >> 16);
    __bf16 out; __builtin_memcpy(&out, &h, 2);
    return out;
}

// ---------------------------------------------------------------------------
// KNN: per-batch pairwise distances over first 3 coords, keep 8 smallest.
// grid = B * (N/256), block = 256 (one query per thread), positions in LDS.
// ---------------------------------------------------------------------------
__global__ __launch_bounds__(256)
void knn_kernel(const float* __restrict__ points, int* __restrict__ srcIdx) {
    __shared__ float px[NPTS], py[NPTS], pz[NPTS];
    const int b    = blockIdx.x >> 3;
    const int tile = blockIdx.x & 7;
    const int tid  = threadIdx.x;
    for (int j = tid; j < NPTS; j += 256) {
        const float* p = points + (size_t)(b * NPTS + j) * 5;
        px[j] = p[0]; py[j] = p[1]; pz[j] = p[2];
    }
    __syncthreads();
    const int q = tile * 256 + tid;
    const float qx = px[q], qy = py[q], qz = pz[q];
    float bestd[8]; int besti[8];
#pragma unroll
    for (int p = 0; p < 8; ++p) { bestd[p] = 3.0e38f; besti[p] = 0; }
    for (int j = 0; j < NPTS; ++j) {
        float dx = px[j] - qx, dy = py[j] - qy, dz = pz[j] - qz;
        float d = dx * dx + dy * dy + dz * dz;
        if (j == q) d = 3.0e38f;                    // exclude self (diag +1e10)
        if (d < bestd[7]) {                          // branchless sorted insert
            int ji = j;
#pragma unroll
            for (int p = 0; p < 8; ++p) {
                bool c  = d < bestd[p];
                float nd = c ? d : bestd[p];
                float od = c ? bestd[p] : d;
                int   ni = c ? ji : besti[p];
                int   oi = c ? besti[p] : ji;
                bestd[p] = nd; d = od; besti[p] = ni; ji = oi;
            }
        }
    }
    const int base = (b * NPTS + q) * KNN;
#pragma unroll
    for (int p = 0; p < 8; ++p) srcIdx[base + p] = b * NPTS + besti[p];
}

// ---------------------------------------------------------------------------
// Pre-convert weights: W[k][c] (f32, row-major) -> Wt[c][kpad] (bf16), zero pad.
// ---------------------------------------------------------------------------
__global__ void prep_w(const float* __restrict__ W, __bf16* __restrict__ Wt,
                       int CF, int KPAD, int COUT) {
    int t = threadIdx.x + blockIdx.x * blockDim.x;
    if (t < COUT * KPAD) {
        int c = t / KPAD;
        int k = t - c * KPAD;
        float v = (k < CF) ? W[k * COUT + c] : 0.0f;
        Wt[t] = f2bf(v);
    }
}

__global__ void zero_kernel(float* p, int n) {
    int i = threadIdx.x + blockIdx.x * blockDim.x;
    if (i < n) p[i] = 0.0f;
}

// ---------------------------------------------------------------------------
// Fused edge-conv GEMM:  Y = lrelu([x_dst, x_src-x_dst] @ W + b)
// then reduce per-dst max/min (8 consecutive edges == one D-fragment half)
// and accumulate per-channel sum/sum^2 for BN.
// Block = 128 threads (4 waves); each block grid-strides over 16-edge M-tiles.
// A 16-edge tile == exactly 2 dst nodes; lane-half of the WMMA D fragment
// holds all 8 edges of one dst, so the segment reduction is lane-local.
// Gather: dst features + src indices staged to LDS once per tile, then the
// per-edge pass reads only x[src] as float4 (B128) vector loads.
// ---------------------------------------------------------------------------
template <int CINF, int KPAD, int COUT>
__global__ __launch_bounds__(128)
void edgeconv_gemm(const float* __restrict__ xin,
                   const int*   __restrict__ srcIdx,
                   const __bf16* __restrict__ Wt,
                   const float* __restrict__ bias,
                   float* __restrict__ ymax,
                   float* __restrict__ ymin,
                   float* __restrict__ stats /* [2*COUT] */) {
    constexpr int CF = 2 * CINF;
    constexpr int KT = KPAD / 32;
    constexpr int NT = COUT / 16;
    constexpr bool VEC = (CINF % 4) == 0;
    constexpr int C4 = VEC ? (CINF / 4) : 1;       // float4 groups per half-row

    __shared__ __align__(16) __bf16 feat[16 * KPAD];
    __shared__ float  dfeat[2 * CINF];             // the 2 dst feature vectors
    __shared__ int    sIdx[16];                    // src index per tile row
    __shared__ float  sstat[2 * COUT];

    const int tid  = threadIdx.x;
    const int wave = tid >> 5;
    const int lane = tid & 31;
    const int half = lane >> 4;
    const int l16  = lane & 15;

    for (int t = tid; t < 2 * COUT; t += 128) sstat[t] = 0.0f;
    __syncthreads();

    for (int mt = blockIdx.x; mt < MTILES; mt += gridDim.x) {
        __syncthreads();
        if constexpr (VEC) {
            // Phase A: stage 2 dst vectors (float4) and 16 src indices.
            for (int t = tid; t < 2 * C4 + 16; t += 128) {
                if (t < 2 * C4) {
                    int dl = t / C4;
                    int c4 = t - dl * C4;
                    const float4 v = *(const float4*)(xin +
                        (size_t)(mt * 2 + dl) * CINF + c4 * 4);
                    float* dst = &dfeat[dl * CINF + c4 * 4];
                    dst[0] = v.x; dst[1] = v.y; dst[2] = v.z; dst[3] = v.w;
                } else {
                    int r = t - 2 * C4;
                    sIdx[r] = srcIdx[mt * 16 + r];
                }
            }
            __syncthreads();
            // Phase B: per-edge src loads (float4) + bf16 LDS fill.
            for (int t = tid; t < 16 * C4; t += 128) {
                int row = t / C4;
                int c4  = t - row * C4;
                int c   = c4 * 4;
                const float4 sv = *(const float4*)(xin +
                    (size_t)sIdx[row] * CINF + c);
                const float* dv = &dfeat[(row >> 3) * CINF + c];
                __bf16* fr = &feat[row * KPAD + c];
                fr[0] = f2bf(dv[0]); fr[1] = f2bf(dv[1]);
                fr[2] = f2bf(dv[2]); fr[3] = f2bf(dv[3]);
                __bf16* fd = fr + CINF;
                fd[0] = f2bf(sv.x - dv[0]); fd[1] = f2bf(sv.y - dv[1]);
                fd[2] = f2bf(sv.z - dv[2]); fd[3] = f2bf(sv.w - dv[3]);
            }
        } else {
            // Scalar path (layer 1: CINF=5, K zero-padded to 32).
            for (int t = tid; t < 16 * KPAD; t += 128) {
                int row = t / KPAD;
                int c   = t - row * KPAD;
                int e   = mt * 16 + row;
                int d   = e >> 3;
                float v = 0.0f;
                if (c < CINF) {
                    v = xin[d * CINF + c];
                } else if (c < CF) {
                    int cc = c - CINF;
                    int s  = srcIdx[e];
                    v = xin[s * CINF + cc] - xin[d * CINF + cc];
                }
                feat[t] = f2bf(v);
            }
        }
        __syncthreads();

        for (int nt = wave; nt < NT; nt += 4) {
            v8f acc = {0.f, 0.f, 0.f, 0.f, 0.f, 0.f, 0.f, 0.f};
            const int col = nt * 16 + l16;
#pragma unroll
            for (int kt = 0; kt < KT; ++kt) {
                // A fragment (16x32 bf16): lane l16 = row M; halves pick K sets.
                const __bf16* fb = &feat[l16 * KPAD + kt * 32 + half * 8];
                v8bf alo = *(const v8bf*)(fb);        // K = base+0..7
                v8bf ahi = *(const v8bf*)(fb + 16);   // K = base+16..23
                v16bf a = __builtin_shufflevector(alo, ahi,
                    0, 1, 2, 3, 4, 5, 6, 7, 8, 9, 10, 11, 12, 13, 14, 15);
                // B fragment (32x16): col fixed per lane, K = base+0..7, base+16..23.
                const __bf16* wb = Wt + col * KPAD + kt * 32 + half * 8;
                v8bf blo = *(const v8bf*)(wb);
                v8bf bhi = *(const v8bf*)(wb + 16);
                v16bf bm = __builtin_shufflevector(blo, bhi,
                    0, 1, 2, 3, 4, 5, 6, 7, 8, 9, 10, 11, 12, 13, 14, 15);
                acc = __builtin_amdgcn_wmma_f32_16x16x32_bf16(
                    false, a, false, bm, (short)0, acc, false, false);
            }
            // Epilogue: bias + leaky-relu, per-dst max/min + BN statistics.
            const float bv = bias[col];
            float mx = -3.0e38f, mn = 3.0e38f, sm = 0.0f, s2 = 0.0f;
#pragma unroll
            for (int v = 0; v < 8; ++v) {
                float y = acc[v] + bv;
                y = (y >= 0.0f) ? y : 0.2f * y;
                mx = fmaxf(mx, y);
                mn = fminf(mn, y);
                sm += y;
                s2 += y * y;
            }
            const int d = mt * 2 + half;             // lane-half == dst node
            ymax[d * COUT + col] = mx;
            ymin[d * COUT + col] = mn;
            atomicAdd(&sstat[col], sm);              // ds_add_f32
            atomicAdd(&sstat[COUT + col], s2);
        }
    }
    __syncthreads();
    for (int t = tid; t < 2 * COUT; t += 128) atomicAdd(&stats[t], sstat[t]);
}

// ---------------------------------------------------------------------------
// BN finalize: mean/var over E edges -> per-channel scale/shift.
// ---------------------------------------------------------------------------
__global__ void bn_finalize(const float* __restrict__ stats,
                            const float* __restrict__ g,
                            const float* __restrict__ be,
                            float* __restrict__ scsh, int C, float invE) {
    int c = threadIdx.x + blockIdx.x * blockDim.x;
    if (c < C) {
        float mean = stats[c] * invE;
        float var  = stats[C + c] * invE - mean * mean;
        float s    = g[c] * rsqrtf(var + 1e-5f);
        scsh[c]     = s;
        scsh[C + c] = be[c] - mean * s;
    }
}

// max(bn(y)) == bn(max y) when scale>=0 else bn(min y)  (affine monotonic).
__global__ void bn_apply_max(const float* __restrict__ ymax,
                             const float* __restrict__ ymin,
                             const float* __restrict__ scsh,
                             float* __restrict__ xout, int Cmask, int C,
                             int total) {
    for (int i = threadIdx.x + blockIdx.x * blockDim.x; i < total;
         i += blockDim.x * gridDim.x) {
        int c = i & Cmask;                           // C is a power of two
        float s = scsh[c];
        float v = (s >= 0.0f) ? ymax[i] : ymin[i];
        xout[i] = v * s + scsh[C + c];
    }
}

// ---------------------------------------------------------------------------
// Global max-pool over N for concat([x1, x2, x3]) -> gfeat (B, 448).
// grid = B*7 (64-channel chunks), block = 512 = 64 channels x 8 n-partitions.
// ---------------------------------------------------------------------------
__global__ __launch_bounds__(512)
void maxpool_kernel(const float* __restrict__ x1, const float* __restrict__ x2,
                    const float* __restrict__ x3, float* __restrict__ gfeat) {
    __shared__ float red[512];
    const int b    = blockIdx.x / 7;
    const int cb   = blockIdx.x % 7;
    const int t    = threadIdx.x;
    const int cl   = t & 63;
    const int part = t >> 6;
    const int c    = cb * 64 + cl;
    const float* src; int C, cc;
    if (c < 64)        { src = x1; C = 64;  cc = c; }
    else if (c < 192)  { src = x2; C = 128; cc = c - 64; }
    else               { src = x3; C = 256; cc = c - 192; }
    float m = -3.0e38f;
    for (int n = part; n < NPTS; n += 8)
        m = fmaxf(m, src[(size_t)(b * NPTS + n) * C + cc]);
    red[t] = m;
    __syncthreads();
    if (part == 0) {
#pragma unroll
        for (int p = 1; p < 8; ++p) m = fmaxf(m, red[p * 64 + cl]);
        gfeat[b * 448 + c] = m;
    }
}

// ---------------------------------------------------------------------------
// Head: out = bn(lrelu(gfeat @ Wf1 + bf1)) @ Wf2 + bf2   (tiny: one block)
// ---------------------------------------------------------------------------
__global__ __launch_bounds__(512)
void head_kernel(const float* __restrict__ gfeat,
                 const float* __restrict__ Wf1, const float* __restrict__ bf1,
                 const float* __restrict__ gf,  const float* __restrict__ bef,
                 const float* __restrict__ Wf2, const float* __restrict__ bf2,
                 float* __restrict__ out) {
    __shared__ float gl[BATCH * 448];
    __shared__ float hl[BATCH * 512];
    const int t = threadIdx.x;
    for (int i = t; i < BATCH * 448; i += 512) gl[i] = gfeat[i];
    __syncthreads();
    {
        const int j = t;                              // 0..511
        float acc[BATCH];
        float bv = bf1[j];
#pragma unroll
        for (int r = 0; r < BATCH; ++r) acc[r] = bv;
        for (int k = 0; k < 448; ++k) {
            float w = Wf1[k * 512 + j];
#pragma unroll
            for (int r = 0; r < BATCH; ++r) acc[r] = fmaf(gl[r * 448 + k], w, acc[r]);
        }
        float sm = 0.0f, s2 = 0.0f;
#pragma unroll
        for (int r = 0; r < BATCH; ++r) {
            float y = acc[r];
            y = (y >= 0.0f) ? y : 0.2f * y;
            acc[r] = y; sm += y; s2 += y * y;
        }
        float mean = sm * (1.0f / BATCH);
        float var  = s2 * (1.0f / BATCH) - mean * mean;
        float s    = gf[j] * rsqrtf(var + 1e-5f);
        float sh   = bef[j] - mean * s;
#pragma unroll
        for (int r = 0; r < BATCH; ++r) hl[r * 512 + j] = fmaf(acc[r], s, sh);
    }
    __syncthreads();
    if (t < 128) {
        float acc[BATCH];
        float bv = bf2[t];
#pragma unroll
        for (int r = 0; r < BATCH; ++r) acc[r] = bv;
        for (int k = 0; k < 512; ++k) {
            float w = Wf2[k * 128 + t];
#pragma unroll
            for (int r = 0; r < BATCH; ++r) acc[r] = fmaf(hl[r * 512 + k], w, acc[r]);
        }
#pragma unroll
        for (int r = 0; r < BATCH; ++r) out[r * 128 + t] = acc[r];
    }
}

// ---------------------------------------------------------------------------
extern "C" void kernel_launch(void* const* d_in, const int* in_sizes, int n_in,
                              void* d_out, int out_size, void* d_ws, size_t ws_size,
                              hipStream_t stream) {
    (void)in_sizes; (void)n_in; (void)out_size; (void)ws_size;
    const float* points = (const float*)d_in[0];
    const float* W1  = (const float*)d_in[1];
    const float* b1  = (const float*)d_in[2];
    const float* g1  = (const float*)d_in[3];
    const float* be1 = (const float*)d_in[4];
    const float* W2  = (const float*)d_in[5];
    const float* b2  = (const float*)d_in[6];
    const float* g2  = (const float*)d_in[7];
    const float* be2 = (const float*)d_in[8];
    const float* W3  = (const float*)d_in[9];
    const float* b3  = (const float*)d_in[10];
    const float* g3  = (const float*)d_in[11];
    const float* be3 = (const float*)d_in[12];
    const float* Wf1 = (const float*)d_in[13];
    const float* bf1 = (const float*)d_in[14];
    const float* gfv = (const float*)d_in[15];
    const float* befv = (const float*)d_in[16];
    const float* Wf2 = (const float*)d_in[17];
    const float* bf2 = (const float*)d_in[18];

    char* ws = (char*)d_ws;
    size_t off = 0;
    auto take = [&](size_t bytes) -> void* {
        void* p = ws + off;
        off = (off + bytes + 255) & ~(size_t)255;
        return p;
    };
    int*    srcIdx = (int*)   take((size_t)NEDGE * 4);
    __bf16* Wt1    = (__bf16*)take((size_t)64 * 32 * 2);
    __bf16* Wt2    = (__bf16*)take((size_t)128 * 128 * 2);
    __bf16* Wt3    = (__bf16*)take((size_t)256 * 256 * 2);
    float*  x1     = (float*) take((size_t)BNPTS * 64 * 4);
    float*  x2     = (float*) take((size_t)BNPTS * 128 * 4);
    float*  x3     = (float*) take((size_t)BNPTS * 256 * 4);
    float*  ymax   = (float*) take((size_t)BNPTS * 256 * 4);
    float*  ymin   = (float*) take((size_t)BNPTS * 256 * 4);
    float*  stats  = (float*) take((size_t)896 * 4);   // [2*64 | 2*128 | 2*256]
    float*  scsh   = (float*) take((size_t)896 * 4);
    float*  gfeat  = (float*) take((size_t)BATCH * 448 * 4);

    float* stats1 = stats;        float* scsh1 = scsh;
    float* stats2 = stats + 128;  float* scsh2 = scsh + 128;
    float* stats3 = stats + 384;  float* scsh3 = scsh + 384;

    const float invE = 1.0f / (float)NEDGE;

    zero_kernel<<<4, 256, 0, stream>>>(stats, 896);
    prep_w<<<(64 * 32 + 255) / 256, 256, 0, stream>>>(W1, Wt1, 10, 32, 64);
    prep_w<<<(128 * 128 + 255) / 256, 256, 0, stream>>>(W2, Wt2, 128, 128, 128);
    prep_w<<<(256 * 256 + 255) / 256, 256, 0, stream>>>(W3, Wt3, 256, 256, 256);
    knn_kernel<<<64, 256, 0, stream>>>(points, srcIdx);

    // Layer 1: CIN=5 (K padded 10->32), COUT=64
    edgeconv_gemm<5, 32, 64><<<2048, 128, 0, stream>>>(
        points, srcIdx, Wt1, b1, ymax, ymin, stats1);
    bn_finalize<<<1, 64, 0, stream>>>(stats1, g1, be1, scsh1, 64, invE);
    bn_apply_max<<<2048, 256, 0, stream>>>(ymax, ymin, scsh1, x1, 63, 64,
                                           BNPTS * 64);

    // Layer 2: CIN=64 (K=128), COUT=128
    edgeconv_gemm<64, 128, 128><<<2048, 128, 0, stream>>>(
        x1, srcIdx, Wt2, b2, ymax, ymin, stats2);
    bn_finalize<<<1, 128, 0, stream>>>(stats2, g2, be2, scsh2, 128, invE);
    bn_apply_max<<<2048, 256, 0, stream>>>(ymax, ymin, scsh2, x2, 127, 128,
                                           BNPTS * 128);

    // Layer 3: CIN=128 (K=256), COUT=256
    edgeconv_gemm<128, 256, 256><<<2048, 128, 0, stream>>>(
        x2, srcIdx, Wt3, b3, ymax, ymin, stats3);
    bn_finalize<<<1, 256, 0, stream>>>(stats3, g3, be3, scsh3, 256, invE);
    bn_apply_max<<<4096, 256, 0, stream>>>(ymax, ymin, scsh3, x3, 255, 256,
                                           BNPTS * 256);

    maxpool_kernel<<<56, 512, 0, stream>>>(x1, x2, x3, gfeat);
    head_kernel<<<1, 512, 0, stream>>>(gfeat, Wf1, bf1, gfv, befv, Wf2, bf2,
                                       (float*)d_out);
}
